// MoEQKVParallelLinear_76811195122155
// MI455X (gfx1250) — matrix operand order repaired and checked by reading, not compiled, not run on hardware
//
#include <hip/hip_runtime.h>

// ---------------------------------------------------------------------------
// MoE QKV projection: out[t] = x[t] @ W[id[t]].T
//   x   : [16384, 2048] f32     (converted once to bf16 in d_ws)
//   W   : [4, 3072, 2048] f32   (converted once to bf16 in d_ws)
//   map : [16384] i32, sorted -> contiguous expert groups
//   out : [16384, 3072] f32
// Strategy: bf16 WMMA (v_wmma_f32_16x16x32_bf16) GEMM, 128x128x32 block
// tiles, 8 waves/block (wave32).
//   - data movement: GLOBAL_LOAD_ASYNC_TO_LDS_B128 + ASYNCcnt, LDS
//     double-buffered with MANUALLY unrolled ping-pong (constant buffer per
//     phase -> ds_load immediate offsets, no register rotation).
//   - expert boundaries: full unmasked GEMM per expert in the block's
//     [eLo,eHi] range; per-row predicated epilogue stores. Inner loop is
//     pure ds_load + wmma.
// ---------------------------------------------------------------------------

typedef __bf16        v16bf __attribute__((ext_vector_type(16)));
typedef float         v8f   __attribute__((ext_vector_type(8)));
typedef unsigned int  v4u   __attribute__((ext_vector_type(4)));
typedef float         v4f   __attribute__((ext_vector_type(4)));
typedef int           v4i   __attribute__((ext_vector_type(4)));

#define NUM_TOKENS  16384
#define HIDDEN      2048
#define QKV_OUT     3072
#define NUM_EXPERTS 4

#define BM 128
#define BN 128
#define BK 32
#define LDK 40                  // padded LDS row stride (bf16): 80B, 16B-aligned
#define KT  (HIDDEN / BK)       // 64 k-steps (even)

#if __has_builtin(__builtin_amdgcn_global_load_async_to_lds_b128) && \
    __has_builtin(__builtin_amdgcn_s_wait_asynccnt)
#define USE_ASYNC 1
#else
#define USE_ASYNC 0
#endif

union FragCast { struct { v4u lo, hi; } q; v16bf v; };

#if USE_ASYNC
// builtin signature (probe-confirmed): (int4 AS1* gsrc, int4 AS3* ldst,
// imm int offset, imm int cpol)
typedef __attribute__((address_space(1))) v4i gv4i;
typedef __attribute__((address_space(3))) v4i lv4i;

__device__ __forceinline__ void async_copy16(const __bf16* g, __bf16* l) {
    __builtin_amdgcn_global_load_async_to_lds_b128(
        (gv4i*)g, (lv4i*)l, /*imm offset*/ 0, /*cpol*/ 0);
}
#endif

// ---- fp32 -> bf16 (RNE) conversion, 8 elems / thread ----------------------
__device__ __forceinline__ unsigned int pk2_bf16(float x, float y) {
    unsigned int ux = __builtin_bit_cast(unsigned int, x);
    unsigned int uy = __builtin_bit_cast(unsigned int, y);
    ux = (ux + 0x7FFFu + ((ux >> 16) & 1u)) >> 16;
    uy = (uy + 0x7FFFu + ((uy >> 16) & 1u)) >> 16;
    return (ux & 0xFFFFu) | (uy << 16);
}

__global__ __launch_bounds__(256)
void f32_to_bf16_vec8(const float* __restrict__ in, __bf16* __restrict__ out,
                      long n8) {
    long i = (long)blockIdx.x * blockDim.x + threadIdx.x;
    if (i >= n8) return;
    v4f a = ((const v4f*)in)[2 * i];
    v4f b = ((const v4f*)in)[2 * i + 1];
    v4u o;
    o.x = pk2_bf16(a.x, a.y);
    o.y = pk2_bf16(a.z, a.w);
    o.z = pk2_bf16(b.x, b.y);
    o.w = pk2_bf16(b.z, b.w);
    ((v4u*)out)[i] = o;
}

// ---- main GEMM ------------------------------------------------------------
__global__ __launch_bounds__(256)
void moe_qkv_bf16_wmma(const __bf16* __restrict__ X,
                       const __bf16* __restrict__ W,
                       const int*    __restrict__ map,
                       float*        __restrict__ out) {
#if USE_ASYNC
    __shared__ __bf16 sA[2][BM * LDK];   // 2 x 10 KB
    __shared__ __bf16 sB[2][BN * LDK];   // 2 x 10 KB
#else
    __shared__ __bf16 sA1[BM * LDK];
    __shared__ __bf16 sB1[BN * LDK];
#endif

    const int t     = threadIdx.x;
    const int lane  = t & 31;
    const int wave  = t >> 5;
    const int waveM = wave >> 2;      // 0..1 -> 64 rows each
    const int waveN = wave & 3;       // 0..3 -> 32 cols each

    const int mBase = blockIdx.y * BM;
    const int nBase = blockIdx.x * BN;

    // staging assignment: 4 threads per row, 8 bf16 (16B) per thread
    const int row0 = t >> 2;          // 0..63
    const int row1 = row0 + 64;       // 64..127
    const int kOff = (t & 3) * 8;     // bf16 element offset within BK

    const int eLo = map[mBase];
    const int eHi = map[mBase + BM - 1];
    const bool multi = (eLo != eHi);

    const __bf16* gA0 = X + (size_t)(mBase + row0) * HIDDEN + kOff;
    const __bf16* gA1 = X + (size_t)(mBase + row1) * HIDDEN + kOff;

    // fragment addressing (wave32 WMMA layouts)
    const int rsel = lane & 15;
    const int khA  = (lane & 16) ? 8 : 0;    // A: K halves split 0..7 / 8..15
    const int khB  = (lane & 16) ? 8 : 0;    // B: K halves split 0..7 / 8..15
    const int mAdd = (lane & 16) ? 8 : 0;    // C/D: lanes 16-31 -> M = v+8

    for (int e = eLo; e <= eHi; ++e) {
        const __bf16* gB0 =
            W + ((size_t)e * QKV_OUT + nBase + row0) * HIDDEN + kOff;
        const __bf16* gB1 = gB0 + (size_t)64 * HIDDEN;

        v8f acc[4][2];
#pragma unroll
        for (int i = 0; i < 4; ++i)
#pragma unroll
            for (int j = 0; j < 2; ++j) acc[i][j] = (v8f)(0.0f);

        // pure ds_load + wmma compute step on one LDS buffer pair
        auto computeTile = [&](const __bf16* bA, const __bf16* bB) {
            v16bf af[4], bfr[2];
#pragma unroll
            for (int i = 0; i < 4; ++i) {
                const __bf16* p = bA + (waveM * 64 + i * 16 + rsel) * LDK + khA;
                FragCast f;
                f.q.lo = *(const v4u*)p;         // K = khA .. khA+7
                f.q.hi = *(const v4u*)(p + 16);  // K = khA+16 .. khA+23
                af[i] = f.v;
            }
#pragma unroll
            for (int j = 0; j < 2; ++j) {
                const __bf16* p = bB + (waveN * 32 + j * 16 + rsel) * LDK + khB;
                FragCast f;
                f.q.lo = *(const v4u*)p;         // K = khB .. khB+7
                f.q.hi = *(const v4u*)(p + 16);  // K = khB+16 .. khB+23
                bfr[j] = f.v;
            }
#pragma unroll
            for (int i = 0; i < 4; ++i)
#pragma unroll
                for (int j = 0; j < 2; ++j)
                    acc[i][j] = __builtin_amdgcn_wmma_f32_16x16x32_bf16(
                        false, af[i], false, bfr[j], (short)0, acc[i][j],
                        false, false);
        };

#if USE_ASYNC
        auto issueTile = [&](int kt, int buf) {
            const int ko = kt * BK;
            async_copy16(gA0 + ko, (__bf16*)&sA[buf][row0 * LDK + kOff]);
            async_copy16(gA1 + ko, (__bf16*)&sA[buf][row1 * LDK + kOff]);
            async_copy16(gB0 + ko, (__bf16*)&sB[buf][row0 * LDK + kOff]);
            async_copy16(gB1 + ko, (__bf16*)&sB[buf][row1 * LDK + kOff]);
        };

        issueTile(0, 0);
        __builtin_amdgcn_s_wait_asynccnt(0);
        __syncthreads();

        // manual ping-pong: constant buffer index per phase
        for (int kt2 = 0; kt2 < KT; kt2 += 2) {
            // phase 0: compute buf0, prefetch kt2+1 -> buf1
            issueTile(kt2 + 1, 1);            // kt2+1 <= 63 always valid
            computeTile(sA[0], sB[0]);
            __builtin_amdgcn_s_wait_asynccnt(0);
            __syncthreads();

            // phase 1: compute buf1, prefetch kt2+2 -> buf0
            if (kt2 + 2 < KT) issueTile(kt2 + 2, 0);
            computeTile(sA[1], sB[1]);
            __builtin_amdgcn_s_wait_asynccnt(0);
            __syncthreads();
        }
#else
        // ---------------- fallback: register-staged sync path -------------
        v4u ra0 = *(const v4u*)(gA0);
        v4u ra1 = *(const v4u*)(gA1);
        v4u rb0 = *(const v4u*)(gB0);
        v4u rb1 = *(const v4u*)(gB1);

        for (int kt = 0; kt < KT; ++kt) {
            __syncthreads();
            *(v4u*)&sA1[row0 * LDK + kOff] = ra0;
            *(v4u*)&sA1[row1 * LDK + kOff] = ra1;
            *(v4u*)&sB1[row0 * LDK + kOff] = rb0;
            *(v4u*)&sB1[row1 * LDK + kOff] = rb1;
            __syncthreads();

            if (kt + 1 < KT) {
                const int ko = (kt + 1) * BK;
                ra0 = *(const v4u*)(gA0 + ko);
                ra1 = *(const v4u*)(gA1 + ko);
                rb0 = *(const v4u*)(gB0 + ko);
                rb1 = *(const v4u*)(gB1 + ko);
            }
            computeTile(sA1, sB1);
        }
        __syncthreads();  // LDS reuse across expert iterations
#endif

        // epilogue for this expert
#pragma unroll
        for (int i = 0; i < 4; ++i) {
#pragma unroll
            for (int j = 0; j < 2; ++j) {
                const int rBase = mBase + waveM * 64 + i * 16 + mAdd;
                float* p = out + (size_t)rBase * QKV_OUT +
                           nBase + waveN * 32 + j * 16 + rsel;
                if (!multi) {  // common case: unconditional stores
#pragma unroll
                    for (int v = 0; v < 8; ++v)
                        p[(size_t)v * QKV_OUT] = acc[i][j][v];
                } else {       // boundary block: keep rows of expert e only
#pragma unroll
                    for (int v = 0; v < 8; ++v)
                        if (map[rBase + v] == e)
                            p[(size_t)v * QKV_OUT] = acc[i][j][v];
                }
            }
        }
    }
}

// ---------------------------------------------------------------------------
extern "C" void kernel_launch(void* const* d_in, const int* in_sizes, int n_in,
                              void* d_out, int out_size, void* d_ws,
                              size_t ws_size, hipStream_t stream) {
    (void)in_sizes; (void)n_in; (void)out_size; (void)ws_size;

    const float* x   = (const float*)d_in[0];
    const float* W   = (const float*)d_in[1];
    const int*   map = (const int*)d_in[2];
    float*       out = (float*)d_out;

    // workspace: bf16 copies of x (64 MB) and W (48 MB) -> both L2-resident
    __bf16* xb = (__bf16*)d_ws;
    __bf16* Wb = xb + (size_t)NUM_TOKENS * HIDDEN;

    const long nx8 = (long)NUM_TOKENS * HIDDEN / 8;            // 4,194,304
    const long nw8 = (long)NUM_EXPERTS * QKV_OUT * HIDDEN / 8; // 3,145,728

    f32_to_bf16_vec8<<<nx8 / 256, 256, 0, stream>>>(x, xb, nx8);
    f32_to_bf16_vec8<<<nw8 / 256, 256, 0, stream>>>(W, Wb, nw8);

    dim3 grid(QKV_OUT / BN, NUM_TOKENS / BM);  // (24, 128)
    moe_qkv_bf16_wmma<<<grid, 256, 0, stream>>>(xb, Wb, map, out);
}
